// EncoderDecoderModel_2705829397051
// MI455X (gfx1250) — matrix operand order repaired and checked by reading, not compiled, not run on hardware
//
#include <hip/hip_runtime.h>
#include <hip/hip_bf16.h>

// ---------------------------------------------------------------------------
// Seq2Seq LSTM encoder/decoder for MI455X (gfx1250, wave32, WMMA + TDM).
// GEMMs: v_wmma_f32_16x16x32_bf16. Logits B-tile staged to LDS by the
// Tensor Data Mover (tensor_load_to_lds) for 4x intra-block reuse.
// ---------------------------------------------------------------------------

typedef __attribute__((ext_vector_type(16))) __bf16       v16bf;
typedef __attribute__((ext_vector_type(8)))  __bf16       v8bf;
typedef __attribute__((ext_vector_type(8)))  float        v8f;
typedef __attribute__((ext_vector_type(4)))  unsigned int u32x4;
typedef __attribute__((ext_vector_type(8)))  unsigned int u32x8;

#define B_     64
#define T_     50
#define E_     256
#define H_     256
#define L_     3
#define NC_    32000
#define STEPS_ 25
#define PAD_   3
#define SOS_   0

#define LROWS   128              // logits rows staged per block
#define LPITCH  528              // LDS row pitch: 512B row + 16B pad (bank rotate)
#define LSMEM   (LROWS * LPITCH) // 67584 B dynamic LDS

__device__ __forceinline__ __bf16 f2bf(float f) {
  unsigned u = __float_as_uint(f);
  u += 0x7fffu + ((u >> 16) & 1u);               // round-to-nearest-even
  unsigned short h = (unsigned short)(u >> 16);
  __bf16 r; __builtin_memcpy(&r, &h, 2); return r;
}
__device__ __forceinline__ float sigmoidf_(float x) { return 1.0f / (1.0f + __expf(-x)); }

__device__ __forceinline__ v8f wmma_bf16(v16bf a, v16bf b, v8f c) {
  return __builtin_amdgcn_wmma_f32_16x16x32_bf16(false, a, false, b, (short)0, c, false, false);
}

// A fragment (16x32 bf16, wave32): lane L<16 holds row M=L, K {k..k+15};
// lane L+16 holds {k+16..k+31}.  koffA = hi?16:0.
__device__ __forceinline__ v16bf load_afrag(const __bf16* __restrict__ A,
                                            int mrow, int k, int koffA) {
  const __bf16* ap = A + mrow * 256 + k + koffA;
  v8bf a0 = *(const v8bf*)(ap);
  v8bf a1 = *(const v8bf*)(ap + 8);
  return __builtin_shufflevector(a0, a1, 0,1,2,3,4,5,6,7,8,9,10,11,12,13,14,15);
}

// ---------------------------------------------------------------------------
// Fused LSTM cell. Grid = 8 blocks (32 hidden units each), 256 threads
// (8 waves: 4 M-tiles x 2 unit-groups). Each wave owns i/f/g/o tiles of the
// same 16 units, so the nonlinearity runs in registers. The two K=256 GEMM
// passes (x*Wih^T then h*Whh^T) run as one 16-iteration software-pipelined
// loop with register double buffering so WMMAs overlap the next loads.
// ---------------------------------------------------------------------------
__global__ __launch_bounds__(256)
void lstm_cell_kernel(const __bf16* __restrict__ Xin,   // [64,256]
                      const __bf16* __restrict__ Hrd,   // [64,256]
                      __bf16*       __restrict__ Hwr,   // [64,256]
                      float*        __restrict__ Cst,   // [64,256]
                      const __bf16* __restrict__ Wih,   // [1024,256]
                      const __bf16* __restrict__ Whh,   // [1024,256]
                      const float*  __restrict__ bias,  // [1024] (bih+bhh)
                      __bf16*       __restrict__ Xout,  // [64,256] unmasked h_new
                      const int*    __restrict__ lengths,
                      int t)
{
  const int lane  = threadIdx.x & 31;
  const int w     = threadIdx.x >> 5;
  const int l15   = lane & 15;
  const int hi    = lane >> 4;
  const int mbase = (w & 3) * 16;
  const int jbase = blockIdx.x * 32 + (w >> 2) * 16;
  const int mrow  = mbase + l15;
  const int koffA = hi ? 16 : 0;
  const int koffB = hi ? 16 : 0;

  const __bf16* As[2] = { Xin, Hrd };
  const __bf16* Ws[2] = { Wih, Whh };

  auto load_iter = [&](int i, v16bf& av, v16bf* bv) {
    const __bf16* A = As[i >> 3];
    const __bf16* W = Ws[i >> 3];
    const int k = (i & 7) * 32;
    av = load_afrag(A, mrow, k, koffA);
#pragma unroll
    for (int g = 0; g < 4; ++g)
      bv[g] = *(const v16bf*)(W + (size_t)(jbase + g * 256 + l15) * 256 + k + koffB);
  };

  v8f acc[4] = {};
  v16bf a_cur, b_cur[4], a_nxt, b_nxt[4];
  load_iter(0, a_cur, b_cur);
#pragma unroll
  for (int i = 0; i < 16; ++i) {
    if (i < 15) load_iter(i + 1, a_nxt, b_nxt);   // prefetch next fragments
#pragma unroll
    for (int g = 0; g < 4; ++g)
      acc[g] = wmma_bf16(a_cur, b_cur[g], acc[g]);
    a_cur = a_nxt;
#pragma unroll
    for (int g = 0; g < 4; ++g) b_cur[g] = b_nxt[g];
  }

  const int u = jbase + l15;
  const float bi  = bias[u];
  const float bff = bias[256 + u];
  const float bg  = bias[512 + u];
  const float bo  = bias[768 + u];

#pragma unroll
  for (int r = 0; r < 8; ++r) {
    const int m = mbase + r + (hi ? 8 : 0);          // batch row (C/D layout)
    float gi = acc[0][r] + bi;
    float gf = acc[1][r] + bff;
    float gg = acc[2][r] + bg;
    float go = acc[3][r] + bo;
    float cp = Cst[m * 256 + u];
    float cn = sigmoidf_(gf) * cp + sigmoidf_(gi) * tanhf(gg);
    float hn = sigmoidf_(go) * tanhf(cn);
    Xout[m * 256 + u] = f2bf(hn);                    // unmasked, feeds next layer
    const bool valid = (lengths == nullptr) || (t < lengths[m]);
    if (valid) {
      Cst[m * 256 + u] = cn;
      Hwr[m * 256 + u] = f2bf(hn);
    } else {
      Hwr[m * 256 + u] = Hrd[m * 256 + u];           // carry old state forward
    }
  }
}

// ---------------------------------------------------------------------------
// Output projection: [64,256] x [256,32000] + bias -> d_out[b,step,0,:].
// Block = 64(M) x 128(N); grid = 250. Wave 0 DMAs the block's 128x256 bf16
// weight tile into LDS via the Tensor Data Mover (2-D D#, padded rows for
// bank rotation); all waves preload A fragments meanwhile, then WMMA off LDS
// with 4x B reuse across the M-tile waves.
// ---------------------------------------------------------------------------
__global__ __launch_bounds__(256)
void logits_kernel(const __bf16* __restrict__ Xin,   // [64,256] = decoder h (L-1)
                   const __bf16* __restrict__ Wl,    // [32000,256] bf16
                   const float*  __restrict__ bl,    // [32000]
                   float*        __restrict__ out,   // [64,25,32000]
                   int step)
{
  extern __shared__ __align__(16) char smraw[];      // LSMEM bytes

  const int lane   = threadIdx.x & 31;
  const int w      = threadIdx.x >> 5;
  const int l15    = lane & 15;
  const int hi     = lane >> 4;
  const int mbase  = (w & 3) * 16;
  const int half   = w >> 2;
  const int nblock = blockIdx.x * 128;
  const int mrow   = mbase + l15;
  const int koffA  = hi ? 16 : 0;
  const int koffB  = hi ? 16 : 0;

  // ---- wave 0: issue TDM load of rows [nblock, nblock+128) x K=256 --------
  if (w == 0) {
    const unsigned long long ga =
        (unsigned long long)(size_t)(Wl + (size_t)nblock * 256);
    const unsigned lds = (unsigned)(size_t)(&smraw[0]);   // low 32 bits = LDS addr

    u32x4 g0;
    g0[0] = 1u;                                           // count=1, user desc
    g0[1] = lds;                                          // lds_addr
    g0[2] = (unsigned)ga;                                 // global_addr[31:0]
    g0[3] = (unsigned)((ga >> 32) & 0x01ffffffu) | (2u << 30); // addr[56:32] | type=2

    u32x8 g1;
    g1[0] = (1u << 16)        // data_size = 2 bytes
          | (1u << 20)        // pad_enable
          | (6u << 22)        // pad_interval: every 128 DWORDs (= 512B row)
          | (3u << 25);       // pad_amount: 4 DWORDs (16B) -> 528B pitch
    g1[1] = (256u   & 0xffffu) << 16;   // tensor_dim0 = 256 (K)
    g1[2] = (32000u & 0xffffu) << 16;   // tensor_dim1 = 32000 (rows)
    g1[3] = (256u   & 0xffffu) << 16;   // tile_dim0 = 256
    g1[4] = 128u;                       // tile_dim1 = 128, tile_dim2 = 0
    g1[5] = 256u;                       // tensor_dim0_stride = 256
    g1[6] = 0u;
    g1[7] = 0u;

    u32x4 gz = { 0u, 0u, 0u, 0u };      // groups 2/3 unused (zero fields)
    asm volatile("tensor_load_to_lds %0, %1, %2, %3"
                 :: "s"(g0), "s"(g1), "s"(gz), "s"(gz)
                 : "memory");
  }

  // ---- all waves: preload the 8 A fragments while the DMA runs ------------
  v16bf av[8];
#pragma unroll
  for (int kk = 0; kk < 8; ++kk)
    av[kk] = load_afrag(Xin, mrow, kk * 32, koffA);

  if (w == 0) __builtin_amdgcn_s_wait_tensorcnt(0);
  __syncthreads();

  // ---- K loop: B fragments from LDS ---------------------------------------
  v8f acc[4] = {};
#pragma unroll
  for (int kk = 0; kk < 8; ++kk) {
#pragma unroll
    for (int g = 0; g < 4; ++g) {
      const int row_local = half * 64 + g * 16 + l15;
      const v16bf bv = *(const v16bf*)(smraw + row_local * LPITCH
                                             + (kk * 32 + koffB) * 2);
      acc[g] = wmma_bf16(av[kk], bv, acc[g]);
    }
  }

  float bb[4];
#pragma unroll
  for (int g = 0; g < 4; ++g) bb[g] = bl[nblock + half * 64 + g * 16 + l15];

#pragma unroll
  for (int r = 0; r < 8; ++r) {
    const int m = mbase + r + (hi ? 8 : 0);
    float* orow = out + ((size_t)m * STEPS_ + step) * NC_;
#pragma unroll
    for (int g = 0; g < 4; ++g)
      orow[nblock + half * 64 + g * 16 + l15] = acc[g][r] + bb[g];
  }
}

// ---------------------------------------------------------------------------
// Greedy argmax over 32000 logits (first-max tie break) fused with the
// next-token embedding gather (E==256==blockDim). One block per batch row.
// ---------------------------------------------------------------------------
__global__ __launch_bounds__(256)
void argmax_embed_kernel(const float* __restrict__ out, int step,
                         const float* __restrict__ dec_embed,
                         __bf16* __restrict__ xdec0)
{
  const int b = blockIdx.x;
  const float* row = out + ((size_t)b * STEPS_ + step) * NC_;
  float best = -3.4e38f; int bidx = 0x7fffffff;
  for (int n = threadIdx.x; n < NC_; n += 256) {  // ascending -> strict > keeps first
    float v = row[n];
    if (v > best) { best = v; bidx = n; }
  }
  __shared__ float sv[256];
  __shared__ int   si[256];
  sv[threadIdx.x] = best; si[threadIdx.x] = bidx;
  __syncthreads();
  for (int s = 128; s > 0; s >>= 1) {
    if (threadIdx.x < s) {
      float v2 = sv[threadIdx.x + s]; int i2 = si[threadIdx.x + s];
      if (v2 > sv[threadIdx.x] || (v2 == sv[threadIdx.x] && i2 < si[threadIdx.x])) {
        sv[threadIdx.x] = v2; si[threadIdx.x] = i2;
      }
    }
    __syncthreads();
  }
  const int tok = si[0];
  xdec0[b * 256 + threadIdx.x] = f2bf(dec_embed[(size_t)tok * 256 + threadIdx.x]);
}

// ---------------------------- small prep kernels ---------------------------
__global__ void cvt_f32_bf16_kernel(const float* __restrict__ in,
                                    __bf16* __restrict__ outp, int n)
{
  for (int i = blockIdx.x * blockDim.x + threadIdx.x; i < n; i += gridDim.x * blockDim.x)
    outp[i] = f2bf(in[i]);
}

__global__ void bias_add_kernel(const float* __restrict__ a,
                                const float* __restrict__ b,
                                float* __restrict__ o, int n)
{
  int i = blockIdx.x * blockDim.x + threadIdx.x;
  if (i < n) o[i] = a[i] + b[i];
}

__global__ void lengths_kernel(const int* __restrict__ x, int* __restrict__ len)
{
  int b = threadIdx.x;
  if (b < B_) {
    int c = 0;
    for (int t = 0; t < T_; ++t) c += (x[b * T_ + t] != PAD_) ? 1 : 0;
    len[b] = c;
  }
}

__global__ void enc_embed_kernel(const int* __restrict__ x,
                                 const float* __restrict__ emb,
                                 __bf16* __restrict__ Xenc)   // [T][B][E]
{
  int idx = blockIdx.x * blockDim.x + threadIdx.x;     // 50*64*256 = 819200
  if (idx < T_ * B_ * E_) {
    int t = idx / (B_ * E_);
    int r = idx % (B_ * E_);
    int b = r / E_, e = r % E_;
    int tok = x[b * T_ + t];
    Xenc[idx] = f2bf(emb[(size_t)tok * E_ + e]);
  }
}

__global__ void init_state_kernel(float* __restrict__ Cst,
                                  __bf16* __restrict__ H0, int n)
{
  for (int i = blockIdx.x * blockDim.x + threadIdx.x; i < n; i += gridDim.x * blockDim.x) {
    Cst[i] = 0.0f;
    H0[i]  = f2bf(0.0f);
  }
}

__global__ void sos_embed_kernel(const float* __restrict__ dec_embed,
                                 __bf16* __restrict__ xdec0)
{
  int b = blockIdx.x;
  xdec0[b * 256 + threadIdx.x] = f2bf(dec_embed[(size_t)SOS_ * E_ + threadIdx.x]);
}

// ---------------------------------------------------------------------------
extern "C" void kernel_launch(void* const* d_in, const int* in_sizes, int n_in,
                              void* d_out, int out_size, void* d_ws, size_t ws_size,
                              hipStream_t stream)
{
  (void)in_sizes; (void)n_in; (void)out_size; (void)ws_size;

  const int*   x         = (const int*)  d_in[0];
  /* d_in[1] target_val unused (teacher forcing ratio == 0) */
  const float* enc_embed = (const float*)d_in[2];
  const float* enc_Wih   = (const float*)d_in[3];
  const float* enc_Whh   = (const float*)d_in[4];
  const float* enc_bih   = (const float*)d_in[5];
  const float* enc_bhh   = (const float*)d_in[6];
  const float* dec_embed = (const float*)d_in[7];
  const float* dec_Wih   = (const float*)d_in[8];
  const float* dec_Whh   = (const float*)d_in[9];
  const float* dec_bih   = (const float*)d_in[10];
  const float* dec_bhh   = (const float*)d_in[11];
  const float* lin_W     = (const float*)d_in[12];
  const float* lin_b     = (const float*)d_in[13];
  float* out = (float*)d_out;

  // ---- workspace carve (~25 MB) -------------------------------------------
  char* base = (char*)d_ws;
  size_t off = 0;
  auto alloc = [&](size_t bytes) -> char* {
    char* p = base + off;
    off = (off + bytes + 511) & ~(size_t)511;
    return p;
  };
  const size_t WN = (size_t)L_ * 4 * H_ * H_;            // 786432 per weight set
  __bf16* encWihB = (__bf16*)alloc(WN * 2);
  __bf16* encWhhB = (__bf16*)alloc(WN * 2);
  __bf16* decWihB = (__bf16*)alloc(WN * 2);
  __bf16* decWhhB = (__bf16*)alloc(WN * 2);
  __bf16* linWB   = (__bf16*)alloc((size_t)NC_ * H_ * 2);
  float*  encBias = (float*) alloc((size_t)L_ * 4 * H_ * 4);
  float*  decBias = (float*) alloc((size_t)L_ * 4 * H_ * 4);
  __bf16* XencB   = (__bf16*)alloc((size_t)T_ * B_ * E_ * 2);
  __bf16* Hbuf    = (__bf16*)alloc((size_t)2 * L_ * B_ * H_ * 2); // ping-pong h
  float*  Cst     = (float*) alloc((size_t)L_ * B_ * H_ * 4);
  __bf16* xb1     = (__bf16*)alloc((size_t)B_ * H_ * 2);
  __bf16* xb2     = (__bf16*)alloc((size_t)B_ * H_ * 2);
  __bf16* xb3     = (__bf16*)alloc((size_t)B_ * H_ * 2);
  __bf16* xdec0   = (__bf16*)alloc((size_t)B_ * E_ * 2);
  int*    lenI    = (int*)   alloc(B_ * 4);

  const int SL = B_ * H_;            // 16384 elems per [64,256] slice
  const int WL = 4 * H_ * H_;        // 262144 elems per layer weight

  // ---- phase 0: weight conversion / bias fold / embeddings / init ---------
  cvt_f32_bf16_kernel<<<2048, 256, 0, stream>>>(enc_Wih, encWihB, (int)WN);
  cvt_f32_bf16_kernel<<<2048, 256, 0, stream>>>(enc_Whh, encWhhB, (int)WN);
  cvt_f32_bf16_kernel<<<2048, 256, 0, stream>>>(dec_Wih, decWihB, (int)WN);
  cvt_f32_bf16_kernel<<<2048, 256, 0, stream>>>(dec_Whh, decWhhB, (int)WN);
  cvt_f32_bf16_kernel<<<2048, 256, 0, stream>>>(lin_W, linWB, NC_ * H_);
  bias_add_kernel<<<12, 256, 0, stream>>>(enc_bih, enc_bhh, encBias, L_ * 4 * H_);
  bias_add_kernel<<<12, 256, 0, stream>>>(dec_bih, dec_bhh, decBias, L_ * 4 * H_);
  lengths_kernel<<<1, 64, 0, stream>>>(x, lenI);
  enc_embed_kernel<<<3200, 256, 0, stream>>>(x, enc_embed, XencB);
  init_state_kernel<<<192, 256, 0, stream>>>(Cst, Hbuf, L_ * B_ * H_); // parity 0

  auto Hp = [&](int par, int l) { return Hbuf + ((size_t)par * L_ + l) * SL; };

  // ---- phase 1: encoder (150 serial cells) --------------------------------
  int par = 0;
  for (int t = 0; t < T_; ++t) {
    for (int l = 0; l < L_; ++l) {
      const __bf16* Xi = (l == 0) ? (XencB + (size_t)t * SL) : ((l == 1) ? xb1 : xb2);
      __bf16*       Xo = (l == 0) ? xb1 : ((l == 1) ? xb2 : xb3);
      lstm_cell_kernel<<<8, 256, 0, stream>>>(
          Xi, Hp(par, l), Hp(par ^ 1, l), Cst + (size_t)l * SL,
          encWihB + (size_t)l * WL, encWhhB + (size_t)l * WL,
          encBias + (size_t)l * 4 * H_, Xo, lenI, t);
    }
    par ^= 1;
  }

  // ---- phase 2: greedy decoder (25 serial steps) --------------------------
  sos_embed_kernel<<<B_, 256, 0, stream>>>(dec_embed, xdec0);
  for (int s = 0; s < STEPS_; ++s) {
    for (int l = 0; l < L_; ++l) {
      const __bf16* Xi = (l == 0) ? xdec0 : ((l == 1) ? xb1 : xb2);
      __bf16*       Xo = (l == 0) ? xb1 : ((l == 1) ? xb2 : xb3);
      lstm_cell_kernel<<<8, 256, 0, stream>>>(
          Xi, Hp(par, l), Hp(par ^ 1, l), Cst + (size_t)l * SL,
          decWihB + (size_t)l * WL, decWhhB + (size_t)l * WL,
          decBias + (size_t)l * 4 * H_, Xo, nullptr, 0);
    }
    par ^= 1;
    logits_kernel<<<NC_ / 128, 256, LSMEM, stream>>>(xb3, linWB, lin_b, out, s);
    argmax_embed_kernel<<<B_, 256, 0, stream>>>(out, s, dec_embed, xdec0);
  }
}
